// MultiHeadAttention_3058016715315
// MI455X (gfx1250) — compile-verified, matrix operand order checked by reading, not run on hardware
//
#include <hip/hip_runtime.h>
#include <hip/hip_bf16.h>
#include <math.h>

// ---------------------------------------------------------------------------
// MultiHeadAttention forward for MI455X (gfx1250, wave32, WMMA f16->f32)
// B=2, S=2048, E=1024, H=16, D=64. Flash-attention style; all WMMA operands
// are contiguous 16-byte runs per lane (b128 loads), and all GEMM loops are
// software-pipelined (register double buffering) so loads overlap WMMAs.
// ---------------------------------------------------------------------------

#define DEV __device__ __forceinline__

typedef __attribute__((ext_vector_type(16))) _Float16 v16h;
typedef __attribute__((ext_vector_type(8)))  _Float16 v8h;
typedef __attribute__((ext_vector_type(8)))  float    v8f;

namespace {
constexpr int kB = 2;
constexpr int kS = 2048;
constexpr int kH = 16;
constexpr int kE = 1024;
constexpr int kD = 64;            // head dim
constexpr float kScale = 0.125f;  // 1/sqrt(64)
}

DEV v8f wmma16(v16h a, v16h b, v8f c) {
  // D = A(16x32 f16) * B(32x16 f16) + C(16x16 f32)
  return __builtin_amdgcn_wmma_f32_16x16x32_f16(
      /*neg_a=*/false, a, /*neg_b=*/false, b,
      /*c_mod=*/(short)0, c, /*reuse_a=*/false, /*reuse_b=*/false);
}

// Uniform fragment loader.  Per ISA 7.12.2, for both A (lane=M) and B
// (lane=N) 16-bit fragments, a lane's 16 f16 values are two contiguous
// 8-element runs at {row_base + half*8} and {row_base + 16 + half*8}.
// Caller passes p = &row_base[half*8]; we load 16B at p and p+16.
DEV v16h load_frag(const _Float16* p) {
  const v8h lo = *(const v8h*)(p);
  const v8h hi = *(const v8h*)(p + 16);
  return __builtin_shufflevector(lo, hi, 0, 1, 2, 3, 4, 5, 6, 7,
                                 8, 9, 10, 11, 12, 13, 14, 15);
}

// ---------------------------------------------------------------------------
// Kernel 0a: plain f32 -> f16 conversion (grid-stride), for x.
// ---------------------------------------------------------------------------
__global__ void __launch_bounds__(256)
cvt_f32_to_f16(const float* __restrict__ src, _Float16* __restrict__ dst, int n) {
  int i = blockIdx.x * blockDim.x + threadIdx.x;
  const int stride = gridDim.x * blockDim.x;
  for (; i < n; i += stride) dst[i] = (_Float16)src[i];
}

// ---------------------------------------------------------------------------
// Kernel 0b: batched f32 -> f16 transpose: src [batch][rows][cols] ->
// dst [batch][cols][rows].  Used for Wq/Wk/Wv ([H,E,D]->[H,D,E]) and
// W_O ([E,E]->[E,E]^T) so B-fragments become contiguous b128 loads.
// ---------------------------------------------------------------------------
__global__ void __launch_bounds__(256)
cvt_transpose(const float* __restrict__ src, _Float16* __restrict__ dst,
              int rows, int cols, int batch) {
  int idx = blockIdx.x * blockDim.x + threadIdx.x;
  const int stride = gridDim.x * blockDim.x;
  const int total = rows * cols * batch;
  for (; idx < total; idx += stride) {
    const int c = idx % cols;
    const int r = (idx / cols) % rows;
    const int bb = idx / (rows * cols);
    dst[((size_t)bb * cols + c) * rows + r] = (_Float16)src[idx];
  }
}

// ---------------------------------------------------------------------------
// Kernel 1: QKV projection.  Block = 128 threads = 4 waves; wave w owns
// D-tile w.  One x A-fragment feeds 3 WMMAs (Q,K,V).  Software-pipelined:
// k-step j+1 fragments are in flight while step j's WMMAs execute.
// Q stored pre-scaled by 1/sqrt(D); V stored transposed [B,H,D,S].
// ---------------------------------------------------------------------------
__global__ void __launch_bounds__(128)
qkv_kernel(const _Float16* __restrict__ xh,
           const _Float16* __restrict__ wqt, const _Float16* __restrict__ wkt,
           const _Float16* __restrict__ wvt,
           const float* __restrict__ bq, const float* __restrict__ bk,
           const float* __restrict__ bv,
           _Float16* __restrict__ qh, _Float16* __restrict__ kh,
           _Float16* __restrict__ vt) {
  const int stile = blockIdx.x % (kS / 16);
  const int bh    = blockIdx.x / (kS / 16);
  const int h = bh % kH, b = bh / kH;
  const int wave = threadIdx.x >> 5;
  const int lane = threadIdx.x & 31;
  const int m = lane & 15, half = lane >> 4;
  const int s0 = stile * 16;
  const int d0 = wave * 16;

  const _Float16* xrow = xh + ((size_t)b * kS + s0 + m) * kE + half * 8;
  const _Float16* wqr = wqt + ((size_t)h * kD + d0 + m) * kE + half * 8;
  const _Float16* wkr = wkt + ((size_t)h * kD + d0 + m) * kE + half * 8;
  const _Float16* wvr = wvt + ((size_t)h * kD + d0 + m) * kE + half * 8;

  v8f cq = {}, ck = {}, cv = {};
  v16h af = load_frag(xrow);
  v16h qf = load_frag(wqr);
  v16h kf = load_frag(wkr);
  v16h vf = load_frag(wvr);
#pragma unroll 2
  for (int k0 = 0; k0 < kE - 32; k0 += 32) {
    const v16h af1 = load_frag(xrow + k0 + 32);
    const v16h qf1 = load_frag(wqr + k0 + 32);
    const v16h kf1 = load_frag(wkr + k0 + 32);
    const v16h vf1 = load_frag(wvr + k0 + 32);
    cq = wmma16(af, qf, cq);
    ck = wmma16(af, kf, ck);
    cv = wmma16(af, vf, cv);
    af = af1; qf = qf1; kf = kf1; vf = vf1;
  }
  cq = wmma16(af, qf, cq);
  ck = wmma16(af, kf, ck);
  cv = wmma16(af, vf, cv);

  const int n = m;  // C layout: lane -> column n, rows i + half*8
  const float biq = bq[h * kD + d0 + n];
  const float bik = bk[h * kD + d0 + n];
  const float biv = bv[h * kD + d0 + n];

  _Float16* qdst = qh + (((size_t)b * kH + h) * kS + s0) * kD + d0;
  _Float16* kdst = kh + (((size_t)b * kH + h) * kS + s0) * kD + d0;
#pragma unroll
  for (int i = 0; i < 8; ++i) {
    const int r = i + half * 8;
    qdst[(size_t)r * kD + n] = (_Float16)((cq[i] + biq) * kScale);
    kdst[(size_t)r * kD + n] = (_Float16)(ck[i] + bik);
  }
  // V transposed: vt[((b*H+h)*D + d)*S + t]; lane writes 8 consecutive t.
  v8h vvec;
#pragma unroll
  for (int i = 0; i < 8; ++i) vvec[i] = (_Float16)(cv[i] + biv);
  *(v8h*)(vt + (((size_t)b * kH + h) * kD + d0 + n) * kS + s0 + half * 8) = vvec;
}

// ---------------------------------------------------------------------------
// Kernel 2: flash attention.  Block = 4 waves; wave w owns query rows
// [blk*64 + 16w, +16).  32 keys/iter: 4 WMMA Q*K^T, causal+padding mask,
// online softmax (width-16 shfl_xor reductions), P via wave-private LDS
// tile, 4 WMMA P*V (V pre-transposed [B,H,D,S]).  The next key block's K^T
// fragments and the current block's V fragments are issued before the
// softmax VALU chain, which hides their latency.
// ---------------------------------------------------------------------------
__global__ void __launch_bounds__(128)
flash_kernel(const _Float16* __restrict__ qh, const _Float16* __restrict__ kh,
             const _Float16* __restrict__ vt,
             const unsigned char* __restrict__ padmask,
             _Float16* __restrict__ ah) {
  const int wave = threadIdx.x >> 5;
  const int lane = threadIdx.x & 31;
  const int m = lane & 15, half = lane >> 4;
  const int h = blockIdx.y % kH, b = blockIdx.y / kH;
  const int qbase = (blockIdx.x * 4 + wave) * 16;

  const _Float16* Qb  = qh + ((size_t)b * kH + h) * kS * kD;
  const _Float16* Kb  = kh + ((size_t)b * kH + h) * kS * kD;
  const _Float16* Vtb = vt + ((size_t)b * kH + h) * kD * kS;
  const unsigned char* pm = padmask + (size_t)b * kS;

  __shared__ alignas(16) _Float16 pbuf[4][16 * 32];
  _Float16* pw = &pbuf[wave][0];

  // Q fragments: two 16x32 A tiles cover D=64 (Q is pre-scaled by 1/8).
  const v16h qa0 = load_frag(Qb + (size_t)(qbase + m) * kD + half * 8);
  const v16h qa1 = load_frag(Qb + (size_t)(qbase + m) * kD + 32 + half * 8);

  v8f o0 = {}, o1 = {}, o2 = {}, o3 = {};
  float rmax[8], rsum[8];
#pragma unroll
  for (int i = 0; i < 8; ++i) { rmax[i] = -INFINITY; rsum[i] = 0.0f; }

  // Preload K^T fragments for the first key block.
  v16h kf0 = load_frag(Kb + (size_t)(0 + m) * kD + half * 8);
  v16h kf1 = load_frag(Kb + (size_t)(0 + m) * kD + 32 + half * 8);
  v16h kf2 = load_frag(Kb + (size_t)(16 + m) * kD + half * 8);
  v16h kf3 = load_frag(Kb + (size_t)(16 + m) * kD + 32 + half * 8);

  const int kend = qbase + 16;  // causal: keys 0 .. qbase+15
  for (int t0 = 0; t0 < kend; t0 += 32) {
    v8f c0 = {}, c1 = {};
    c0 = wmma16(qa0, kf0, c0);
    c0 = wmma16(qa1, kf1, c0);
    c1 = wmma16(qa0, kf2, c1);
    c1 = wmma16(qa1, kf3, c1);

    // Issue V fragments (current block) and K^T fragments (next block,
    // clamped on the last iteration) before the softmax chain.
    const v16h vf0 = load_frag(Vtb + (size_t)(0  + m) * kS + t0 + half * 8);
    const v16h vf1 = load_frag(Vtb + (size_t)(16 + m) * kS + t0 + half * 8);
    const v16h vf2 = load_frag(Vtb + (size_t)(32 + m) * kS + t0 + half * 8);
    const v16h vf3 = load_frag(Vtb + (size_t)(48 + m) * kS + t0 + half * 8);
    const int tn = (t0 + 32 < kend) ? (t0 + 32) : t0;
    kf0 = load_frag(Kb + (size_t)(tn + m) * kD + half * 8);
    kf1 = load_frag(Kb + (size_t)(tn + m) * kD + 32 + half * 8);
    kf2 = load_frag(Kb + (size_t)(tn + 16 + m) * kD + half * 8);
    kf3 = load_frag(Kb + (size_t)(tn + 16 + m) * kD + 32 + half * 8);

    const int key0 = t0 + m, key1 = t0 + 16 + m;
    const bool pad0 = pm[key0] != 0;
    const bool pad1 = pm[key1] != 0;

#pragma unroll
    for (int i = 0; i < 8; ++i) {
      const int qrow = qbase + i + half * 8;
      float s0 = c0[i];
      float s1 = c1[i];
      if (key0 > qrow || pad0) s0 = -INFINITY;
      if (key1 > qrow || pad1) s1 = -INFINITY;
      // online softmax update for row (i + half*8)
      float mx = fmaxf(s0, s1);
      mx = fmaxf(mx, __shfl_xor(mx, 1, 16));
      mx = fmaxf(mx, __shfl_xor(mx, 2, 16));
      mx = fmaxf(mx, __shfl_xor(mx, 4, 16));
      mx = fmaxf(mx, __shfl_xor(mx, 8, 16));
      const float nm = fmaxf(rmax[i], mx);
      const float corr = __expf(rmax[i] - nm);  // 0 on first block
      rmax[i] = nm;
      const float p0 = __expf(s0 - nm);
      const float p1 = __expf(s1 - nm);
      c0[i] = p0;
      c1[i] = p1;
      float rs = p0 + p1;
      rs += __shfl_xor(rs, 1, 16);
      rs += __shfl_xor(rs, 2, 16);
      rs += __shfl_xor(rs, 4, 16);
      rs += __shfl_xor(rs, 8, 16);
      rsum[i] = rsum[i] * corr + rs;
      o0[i] *= corr; o1[i] *= corr; o2[i] *= corr; o3[i] *= corr;
    }

    // C-layout P -> LDS -> A-layout fragment (wave-private tile, no barrier;
    // same-wave LDS RAW resolved by compiler-inserted DScnt waits).
#pragma unroll
    for (int i = 0; i < 8; ++i) {
      const int r = i + half * 8;
      pw[r * 32 + m]      = (_Float16)c0[i];
      pw[r * 32 + 16 + m] = (_Float16)c1[i];
    }
    const v16h pa = load_frag(pw + m * 32 + half * 8);

    // O += P (16x32) * V (32x64)
    o0 = wmma16(pa, vf0, o0);
    o1 = wmma16(pa, vf1, o1);
    o2 = wmma16(pa, vf2, o2);
    o3 = wmma16(pa, vf3, o3);
  }

  // normalize and store into concat buffer [B,S,E] at column h*D
  _Float16* dst = ah + ((size_t)b * kS + qbase) * kE + h * kD;
#pragma unroll
  for (int i = 0; i < 8; ++i) {
    const float inv = 1.0f / rsum[i];
    const int r = i + half * 8;
    dst[(size_t)r * kE + 0  + m] = (_Float16)(o0[i] * inv);
    dst[(size_t)r * kE + 16 + m] = (_Float16)(o1[i] * inv);
    dst[(size_t)r * kE + 32 + m] = (_Float16)(o2[i] * inv);
    dst[(size_t)r * kE + 48 + m] = (_Float16)(o3[i] * inv);
  }
}

// ---------------------------------------------------------------------------
// Kernel 3: output projection [B*S, E] x [E, E] -> f32 d_out.
// W_O pre-transposed; each wave computes FOUR 16x16 N-tiles per A-fragment
// (1 A load + 4 B loads -> 4 WMMAs per k-step), software-pipelined.
// Block = 8 waves; grid = (B*S/16, E/(16*8*4)).
// ---------------------------------------------------------------------------
__global__ void __launch_bounds__(256)
proj_kernel(const _Float16* __restrict__ ah, const _Float16* __restrict__ wot,
            float* __restrict__ out) {
  const int wave = threadIdx.x >> 5;
  const int lane = threadIdx.x & 31;
  const int m = lane & 15, half = lane >> 4;
  const int mtile = blockIdx.x;
  const int ntile = (blockIdx.y * 8 + wave) * 4;  // 4 consecutive N-tiles

  const _Float16* Ar = ah + ((size_t)mtile * 16 + m) * kE + half * 8;
  const _Float16* Br0 = wot + ((size_t)(ntile + 0) * 16 + m) * kE + half * 8;
  const _Float16* Br1 = wot + ((size_t)(ntile + 1) * 16 + m) * kE + half * 8;
  const _Float16* Br2 = wot + ((size_t)(ntile + 2) * 16 + m) * kE + half * 8;
  const _Float16* Br3 = wot + ((size_t)(ntile + 3) * 16 + m) * kE + half * 8;

  v8f c0 = {}, c1 = {}, c2 = {}, c3 = {};
  v16h af = load_frag(Ar);
  v16h b0 = load_frag(Br0);
  v16h b1 = load_frag(Br1);
  v16h b2 = load_frag(Br2);
  v16h b3 = load_frag(Br3);
#pragma unroll 2
  for (int k0 = 0; k0 < kE - 32; k0 += 32) {
    const v16h afn = load_frag(Ar + k0 + 32);
    const v16h b0n = load_frag(Br0 + k0 + 32);
    const v16h b1n = load_frag(Br1 + k0 + 32);
    const v16h b2n = load_frag(Br2 + k0 + 32);
    const v16h b3n = load_frag(Br3 + k0 + 32);
    c0 = wmma16(af, b0, c0);
    c1 = wmma16(af, b1, c1);
    c2 = wmma16(af, b2, c2);
    c3 = wmma16(af, b3, c3);
    af = afn; b0 = b0n; b1 = b1n; b2 = b2n; b3 = b3n;
  }
  c0 = wmma16(af, b0, c0);
  c1 = wmma16(af, b1, c1);
  c2 = wmma16(af, b2, c2);
  c3 = wmma16(af, b3, c3);

  float* dst = out + (size_t)mtile * 16 * kE + ntile * 16;
#pragma unroll
  for (int i = 0; i < 8; ++i) {
    const size_t row = (size_t)(i + half * 8) * kE;
    dst[row + 0  + m] = c0[i];
    dst[row + 16 + m] = c1[i];
    dst[row + 32 + m] = c2[i];
    dst[row + 48 + m] = c3[i];
  }
}

// ---------------------------------------------------------------------------
// Host-side launcher
// ---------------------------------------------------------------------------
extern "C" void kernel_launch(void* const* d_in, const int* in_sizes, int n_in,
                              void* d_out, int out_size, void* d_ws, size_t ws_size,
                              hipStream_t stream) {
  (void)in_sizes; (void)n_in; (void)out_size; (void)ws_size;
  const float* x  = (const float*)d_in[0];
  const float* Wq = (const float*)d_in[1];
  const float* bq = (const float*)d_in[2];
  const float* Wk = (const float*)d_in[3];
  const float* bk = (const float*)d_in[4];
  const float* Wv = (const float*)d_in[5];
  const float* bv = (const float*)d_in[6];
  const float* WO = (const float*)d_in[7];
  // d_in[8] = causal_mask (tril) -- applied analytically (key > query -> -inf)
  const unsigned char* pad = (const unsigned char*)d_in[9];
  float* out = (float*)d_out;

  char* ws = (char*)d_ws;
  size_t off = 0;
  auto alloc_h = [&](size_t elems) {
    _Float16* p = (_Float16*)(ws + off);
    off += elems * sizeof(_Float16);
    return p;
  };
  _Float16* xh  = alloc_h((size_t)kB * kS * kE);      // x f16; reused as concat output
  _Float16* wqh = alloc_h((size_t)kH * kD * kE);      // transposed [H,D,E]
  _Float16* wkh = alloc_h((size_t)kH * kD * kE);
  _Float16* wvh = alloc_h((size_t)kH * kD * kE);
  _Float16* woh = alloc_h((size_t)kE * kE);           // W_O^T
  _Float16* qh  = alloc_h((size_t)kB * kH * kS * kD); // [B,H,S,D], pre-scaled
  _Float16* kh  = alloc_h((size_t)kB * kH * kS * kD); // [B,H,S,D]
  _Float16* vth = alloc_h((size_t)kB * kH * kD * kS); // [B,H,D,S] (transposed)
  _Float16* ah  = xh;  // x is dead after qkv_kernel; alias for the concat buffer

  cvt_f32_to_f16<<<2048, 256, 0, stream>>>(x, xh, kB * kS * kE);
  cvt_transpose<<<1024, 256, 0, stream>>>(Wq, wqh, kE, kD, kH);
  cvt_transpose<<<1024, 256, 0, stream>>>(Wk, wkh, kE, kD, kH);
  cvt_transpose<<<1024, 256, 0, stream>>>(Wv, wvh, kE, kD, kH);
  cvt_transpose<<<1024, 256, 0, stream>>>(WO, woh, kE, kE, 1);

  qkv_kernel<<<kB * kH * (kS / 16), 128, 0, stream>>>(
      xh, wqh, wkh, wvh, bq, bk, bv, qh, kh, vth);

  flash_kernel<<<dim3(kS / 64, kB * kH), 128, 0, stream>>>(qh, kh, vth, pad, ah);

  proj_kernel<<<dim3(kB * kS / 16, kE / (16 * 8 * 4)), 256, 0, stream>>>(ah, woh, out);
}
